// MetricMultiHeadAttention_90039694393582
// MI455X (gfx1250) — matrix operand order repaired and checked by reading, not compiled, not run on hardware
//
#include <hip/hip_runtime.h>
#include <hip/hip_bf16.h>
#include <stdint.h>

// Problem dims (fixed by the reference)
#define Bb 2
#define Ss 2048
#define DMm 512
#define Hh 8
#define DHh 64

typedef __attribute__((ext_vector_type(16))) __bf16 v16bf;
typedef __attribute__((ext_vector_type(8)))  __bf16 bf16x8;
typedef __attribute__((ext_vector_type(8)))  float  v8f;

// ---------- helpers ----------
__device__ __forceinline__ __bf16 f2bf(float f) {
    union { float f; uint32_t u; } c; c.f = f;
    uint32_t r = (c.u + 0x7FFFu + ((c.u >> 16) & 1u)) >> 16;
    union { unsigned short s; __bf16 b; } o; o.s = (unsigned short)r;
    return o.b;
}

// A fragment (16x32 bf16): lane<16 holds row lane, K {0..7, 16..23};
// lane>=16 holds row lane-16, K {8..15, 24..31}.  p = row base at K offset 0.
__device__ __forceinline__ v16bf frag_a16(const __bf16* p, int lane) {
    int hb = (lane >> 4) << 3;                       // 0 or 8
    bf16x8 lo = *(const bf16x8*)(p + hb);
    bf16x8 hi = *(const bf16x8*)(p + 16 + hb);
    v16bf r;
#pragma unroll
    for (int i = 0; i < 8; ++i) { r[i] = lo[i]; r[i + 8] = hi[i]; }
    return r;
}

// B fragment (32x16 bf16): lane<16 holds col lane, K {0..7, 16..23};
// lane>=16 holds col lane-16, K {8..15, 24..31}.  p = col base at K offset 0
// (data laid out contiguous along K, i.e. "transposed weight" rows).
__device__ __forceinline__ v16bf frag_b16(const __bf16* p, int lane) {
    int hb = (lane >> 4) << 3;                       // 0 or 8
    bf16x8 lo = *(const bf16x8*)(p + hb);
    bf16x8 hi = *(const bf16x8*)(p + 16 + hb);
    v16bf r;
#pragma unroll
    for (int i = 0; i < 8; ++i) { r[i] = lo[i]; r[i + 8] = hi[i]; }
    return r;
}

__device__ __forceinline__ v8f wmma_bf16(v16bf a, v16bf b, v8f c) {
    return __builtin_amdgcn_wmma_f32_16x16x32_bf16(
        /*neg_a=*/false, a, /*neg_b=*/false, b,
        /*c_mod=*/(short)0, c, /*reuse_a=*/false, /*reuse_b=*/false);
}

// ---------- tiny conversion kernels ----------
__global__ void cvt_bf16_kernel(const float* __restrict__ in, __bf16* __restrict__ out, int n) {
    int i = blockIdx.x * blockDim.x + threadIdx.x;
    if (i < n) out[i] = f2bf(in[i]);
}

// in: (R,C) row-major fp32  ->  out: (C,R) row-major bf16 (transposed)
__global__ void cvt_bf16_T_kernel(const float* __restrict__ in, __bf16* __restrict__ out, int R, int C) {
    int i = blockIdx.x * blockDim.x + threadIdx.x;
    if (i < R * C) {
        int r = i / C, c = i % C;
        out[c * R + r] = f2bf(in[i]);
    }
}

// Wfused[k][h] = sum_j Wq[k][j] * Wa[j][h];  beff[h] = ba[h] + sum_j bq[j]*Wa[j][h]
__global__ void fuse_wab_kernel(const float* __restrict__ Wq, const float* __restrict__ Wa,
                                const float* __restrict__ ba, const float* __restrict__ bq,
                                float* __restrict__ Wfused, float* __restrict__ beff) {
    int i = blockIdx.x * blockDim.x + threadIdx.x;   // over DM*H
    if (i < DMm * Hh) {
        int k = i / Hh, hh = i % Hh;
        float s = 0.f;
        for (int j = 0; j < DMm; ++j) s += Wq[k * DMm + j] * Wa[j * Hh + hh];
        Wfused[i] = s;
        if (i < Hh) {
            float t = ba[i];
            for (int j = 0; j < DMm; ++j) t += bq[j] * Wa[j * Hh + i];
            beff[i] = t;
        }
    }
}

// coef[b*S+s][h] = q[b,s,:] . Wfused[:,h] + beff[h]   (fp32, high precision path)
__global__ void coef_kernel(const float* __restrict__ q, const float* __restrict__ Wfused,
                            const float* __restrict__ beff, float* __restrict__ coef) {
    int i = blockIdx.x * blockDim.x + threadIdx.x;   // over B*S*H
    if (i < Bb * Ss * Hh) {
        int row = i / Hh, hh = i % Hh;
        float s = beff[hh];
        const float* qr = q + (size_t)row * DMm;
        for (int j = 0; j < DMm; ++j) s += qr[j] * Wfused[j * Hh + hh];
        coef[i] = s;
    }
}

// ---------- WMMA GEMM:  out = A(MxK,bf16) @ Wt(NxK,bf16)^T + bias ----------
// 256 threads = 8 waves, waves arranged 2(M) x 4(N), each wave 32x32 tile.
// Workgroup tile: 64(M) x 128(N).  Direct global loads (weights live in L2: 192MB).
__global__ void __launch_bounds__(256)
gemm_bf16_wmma(const __bf16* __restrict__ A, const __bf16* __restrict__ Wt,
               const float* __restrict__ bias,
               float* __restrict__ outF, __bf16* __restrict__ outB,
               int M, int N, int K, int transS) {
    int lane = threadIdx.x & 31, w = threadIdx.x >> 5;
    int wm = w >> 2, wn = w & 3;
    int m0 = blockIdx.y * 64 + wm * 32;
    int n0 = blockIdx.x * 128 + wn * 32;
    int r = lane & 15;

    v8f acc[2][2] = {};
    const __bf16* pa0 = A  + (size_t)(m0 + r) * K;
    const __bf16* pa1 = A  + (size_t)(m0 + 16 + r) * K;
    const __bf16* pb0 = Wt + (size_t)(n0 + r) * K;
    const __bf16* pb1 = Wt + (size_t)(n0 + 16 + r) * K;

    for (int kk = 0; kk < K; kk += 32) {
        v16bf a0 = frag_a16(pa0 + kk, lane);
        v16bf a1 = frag_a16(pa1 + kk, lane);
        v16bf b0 = frag_b16(pb0 + kk, lane);
        v16bf b1 = frag_b16(pb1 + kk, lane);
        acc[0][0] = wmma_bf16(a0, b0, acc[0][0]);
        acc[0][1] = wmma_bf16(a0, b1, acc[0][1]);
        acc[1][0] = wmma_bf16(a1, b0, acc[1][0]);
        acc[1][1] = wmma_bf16(a1, b1, acc[1][1]);
    }
#pragma unroll
    for (int sm = 0; sm < 2; ++sm)
#pragma unroll
        for (int sn = 0; sn < 2; ++sn)
#pragma unroll
            for (int i = 0; i < 8; ++i) {
                int row = m0 + sm * 16 + i + ((lane >> 4) << 3);
                int col = n0 + sn * 16 + (lane & 15);
                float v = acc[sm][sn][i] + bias[col];
                if (outF) outF[(size_t)row * N + col] = v;
                if (outB) {
                    if (transS > 0) {
                        int bb = row / transS, ss = row % transS;
                        outB[((size_t)bb * N + col) * transS + ss] = f2bf(v);   // (B, DM, S)
                    } else {
                        outB[(size_t)row * N + col] = f2bf(v);
                    }
                }
            }
}

// ---------- fused metric attention ----------
// grid (S/16, H, B), block 256 (8 waves).  LDS holds the full 16 x 2048 fp32
// logit block (128KB < 320KB/WGP) so attn is written to HBM exactly once.
#define LSTR 2052   // 2048 + pad
__global__ void __launch_bounds__(256)
attn_kernel(const __bf16* __restrict__ qp, const __bf16* __restrict__ kp,
            const __bf16* __restrict__ vt,                       // (B, DM, S) key-major
            const float* __restrict__ acoef, const float* __restrict__ bcoef,  // (B*S, H)
            const float* __restrict__ xdiff, const int* __restrict__ mask,
            float* __restrict__ attn_out, __bf16* __restrict__ out_bf) {
    extern __shared__ float smem[];
    float* Lg  = smem;                 // 16 * LSTR
    float* red = smem + 16 * LSTR;     // 4 * 256 partial-out reduce

    int lane = threadIdx.x & 31, w = threadIdx.x >> 5;
    int qt = blockIdx.x, h = blockIdx.y, b = blockIdx.z;
    int q0 = qt * 16;
    int r = lane & 15;
    int hb = (lane >> 4) << 3;

    // q-tile A fragments (16 rows x DH=64, two K=32 slabs), reused all phase 1
    const __bf16* qrow = qp + ((size_t)(b * Ss + q0 + r)) * DMm + h * DHh;
    v16bf aq0 = frag_a16(qrow, lane);
    v16bf aq1 = frag_a16(qrow + 32, lane);

    // ---- phase 1: logits = qk/sqrt(dh) + mask*-1e9 + a*xd + b*xd^2 -> LDS ----
    for (int j = w; j < Ss / 16; j += 8) {
        const __bf16* krow = kp + ((size_t)(b * Ss + j * 16 + r)) * DMm + h * DHh;
        if (j + 8 < Ss / 16)
            __builtin_prefetch(kp + ((size_t)(b * Ss + (j + 8) * 16 + r)) * DMm + h * DHh, 0, 1);
        v16bf bk0 = frag_b16(krow, lane);
        v16bf bk1 = frag_b16(krow + 32, lane);
        v8f c = {};
        c = wmma_bf16(aq0, bk0, c);
        c = wmma_bf16(aq1, bk1, c);

        int key = j * 16 + (lane & 15);
        float mval = (float)mask[b * Ss + key] * -1.0e9f;
#pragma unroll
        for (int i = 0; i < 8; ++i) {
            int rr = i + ((lane >> 4) << 3);
            int qg = q0 + rr;
            float xd = xdiff[((size_t)(b * Ss + qg)) * Ss + key];
            float ac = acoef[(size_t)(b * Ss + qg) * Hh + h];
            float bc = bcoef[(size_t)(b * Ss + qg) * Hh + h];
            Lg[rr * LSTR + key] = c[i] * 0.125f + mval + (ac + bc * xd) * xd;
        }
    }
    __syncthreads();

    // ---- phase 2: softmax per row, write attn once ----
    for (int rr = w * 2; rr < w * 2 + 2; ++rr) {
        float* row = Lg + rr * LSTR;
        float m = -3.0e38f;
        for (int k = lane; k < Ss; k += 32) m = fmaxf(m, row[k]);
#pragma unroll
        for (int off = 16; off >= 1; off >>= 1) m = fmaxf(m, __shfl_xor(m, off, 32));
        float sum = 0.f;
        for (int k = lane; k < Ss; k += 32) { float e = __expf(row[k] - m); row[k] = e; sum += e; }
#pragma unroll
        for (int off = 16; off >= 1; off >>= 1) sum += __shfl_xor(sum, off, 32);
        float inv = 1.0f / sum;
        float* arow = attn_out + (((size_t)(b * Hh + h) * Ss) + q0 + rr) * Ss;
        for (int k = lane; k < Ss; k += 32) { float v = row[k] * inv; row[k] = v; arow[k] = v; }
    }
    for (int i = threadIdx.x; i < 4 * 256; i += 256) red[i] = 0.f;
    __syncthreads();

    // ---- phase 3: out(16x64) = attn(16x2048) @ V(2048x64), WMMA over keys ----
    int nt = w & 3;                 // which 16-wide dh tile
    int kh = w >> 2;                // key half (0..1023 / 1024..2047)
    const __bf16* vbase = vt + ((size_t)b * DMm + h * DHh + nt * 16 + (lane & 15)) * Ss;
    v8f acc = {};
    for (int kb = kh * 1024; kb < kh * 1024 + 1024; kb += 32) {
        v16bf af;
        {
            const float* lr = Lg + (lane & 15) * LSTR + kb;
#pragma unroll
            for (int i = 0; i < 8; ++i) {
                af[i]     = f2bf(lr[hb + i]);
                af[i + 8] = f2bf(lr[16 + hb + i]);
            }
        }
        v16bf bv = frag_b16(vbase + kb, lane);
        acc = wmma_bf16(af, bv, acc);
    }
#pragma unroll
    for (int i = 0; i < 8; ++i) {
        int rr = i + ((lane >> 4) << 3);
        atomicAdd(&red[nt * 256 + rr * 16 + (lane & 15)], acc[i]);
    }
    __syncthreads();
    for (int i = threadIdx.x; i < 1024; i += 256) {
        int nt2 = i >> 8, rr = (i >> 4) & 15, cc = i & 15;
        out_bf[((size_t)(b * Ss + q0 + rr)) * DMm + h * DHh + nt2 * 16 + cc] = f2bf(red[i]);
    }
}

// ---------- host side ----------
static inline size_t alignup(size_t x) { return (x + 255) & ~(size_t)255; }

extern "C" void kernel_launch(void* const* d_in, const int* in_sizes, int n_in,
                              void* d_out, int out_size, void* d_ws, size_t ws_size,
                              hipStream_t stream) {
    const float* q     = (const float*)d_in[0];
    const float* k     = (const float*)d_in[1];
    const float* v     = (const float*)d_in[2];
    const float* xdiff = (const float*)d_in[3];
    const int*   mask  = (const int*)d_in[4];
    const float* Wq = (const float*)d_in[5];  const float* bq = (const float*)d_in[6];
    const float* Wk = (const float*)d_in[7];  const float* bk = (const float*)d_in[8];
    const float* Wv = (const float*)d_in[9];  const float* bv = (const float*)d_in[10];
    const float* Wa = (const float*)d_in[11]; const float* ba = (const float*)d_in[12];
    const float* Wb = (const float*)d_in[13]; const float* bb = (const float*)d_in[14];
    const float* Wo = (const float*)d_in[15]; const float* bo = (const float*)d_in[16];

    const int M = Bb * Ss;                       // 4096 rows
    float*  out_final = (float*)d_out;                         // (B,S,DM)
    float*  attn_out  = (float*)d_out + (size_t)M * DMm;       // (B,H,S,S)

    char* ws = (char*)d_ws;
    size_t off = 0;
    auto take = [&](size_t bytes) { char* p = ws + off; off += alignup(bytes); return p; };
    __bf16* q_bf   = (__bf16*)take((size_t)M * DMm * 2);
    __bf16* k_bf   = (__bf16*)take((size_t)M * DMm * 2);
    __bf16* v_bf   = (__bf16*)take((size_t)M * DMm * 2);
    __bf16* Wq_t   = (__bf16*)take((size_t)DMm * DMm * 2);
    __bf16* Wk_t   = (__bf16*)take((size_t)DMm * DMm * 2);
    __bf16* Wv_t   = (__bf16*)take((size_t)DMm * DMm * 2);
    __bf16* Wo_t   = (__bf16*)take((size_t)DMm * DMm * 2);
    __bf16* qp_bf  = (__bf16*)take((size_t)M * DMm * 2);
    __bf16* kp_bf  = (__bf16*)take((size_t)M * DMm * 2);
    __bf16* vt_bf  = (__bf16*)take((size_t)M * DMm * 2);       // (B, DM, S)
    __bf16* oh_bf  = (__bf16*)take((size_t)M * DMm * 2);       // head-concat attn output
    float*  Wqa    = (float*)take((size_t)DMm * Hh * 4);
    float*  Wqb    = (float*)take((size_t)DMm * Hh * 4);
    float*  ba_eff = (float*)take((size_t)Hh * 4);
    float*  bb_eff = (float*)take((size_t)Hh * 4);
    float*  acoef  = (float*)take((size_t)M * Hh * 4);
    float*  bcoef  = (float*)take((size_t)M * Hh * 4);
    (void)off; (void)ws_size; (void)n_in; (void)in_sizes; (void)out_size;

    const int nElem = M * DMm;                   // 2,097,152
    // 1) fp32 -> bf16 activations; weights -> transposed bf16 (N-major, K contiguous)
    cvt_bf16_kernel<<<(nElem + 255) / 256, 256, 0, stream>>>(q, q_bf, nElem);
    cvt_bf16_kernel<<<(nElem + 255) / 256, 256, 0, stream>>>(k, k_bf, nElem);
    cvt_bf16_kernel<<<(nElem + 255) / 256, 256, 0, stream>>>(v, v_bf, nElem);
    const int nW = DMm * DMm;
    cvt_bf16_T_kernel<<<(nW + 255) / 256, 256, 0, stream>>>(Wq, Wq_t, DMm, DMm);
    cvt_bf16_T_kernel<<<(nW + 255) / 256, 256, 0, stream>>>(Wk, Wk_t, DMm, DMm);
    cvt_bf16_T_kernel<<<(nW + 255) / 256, 256, 0, stream>>>(Wv, Wv_t, DMm, DMm);
    cvt_bf16_T_kernel<<<(nW + 255) / 256, 256, 0, stream>>>(Wo, Wo_t, DMm, DMm);

    // 2) fused metric-coefficient weights:  a = q@(Wq@Wa) + (bq@Wa + ba)
    fuse_wab_kernel<<<(DMm * Hh + 255) / 256, 256, 0, stream>>>(Wq, Wa, ba, bq, Wqa, ba_eff);
    fuse_wab_kernel<<<(DMm * Hh + 255) / 256, 256, 0, stream>>>(Wq, Wb, bb, bq, Wqb, bb_eff);
    coef_kernel<<<(M * Hh + 255) / 256, 256, 0, stream>>>(q, Wqa, ba_eff, acoef);
    coef_kernel<<<(M * Hh + 255) / 256, 256, 0, stream>>>(q, Wqb, bb_eff, bcoef);

    // 3) WMMA projections: qp, kp row-major bf16; vp stored transposed (B,DM,S)
    dim3 ggrid(DMm / 128, M / 64);
    gemm_bf16_wmma<<<ggrid, 256, 0, stream>>>(q_bf, Wq_t, bq, nullptr, qp_bf, M, DMm, DMm, 0);
    gemm_bf16_wmma<<<ggrid, 256, 0, stream>>>(k_bf, Wk_t, bk, nullptr, kp_bf, M, DMm, DMm, 0);
    gemm_bf16_wmma<<<ggrid, 256, 0, stream>>>(v_bf, Wv_t, bv, nullptr, vt_bf, M, DMm, DMm, Ss);

    // 4) fused attention: LDS-resident logits, single attn write, WMMA attn@V
    dim3 agrid(Ss / 16, Hh, Bb);
    size_t smem = (16 * LSTR + 4 * 256) * sizeof(float);       // ~135 KB < 320 KB/WGP
    attn_kernel<<<agrid, 256, smem, stream>>>(qp_bf, kp_bf, vt_bf, acoef, bcoef,
                                              xdiff, mask, attn_out, oh_bf);

    // 5) output projection -> fp32 d_out
    gemm_bf16_wmma<<<ggrid, 256, 0, stream>>>(oh_bf, Wo_t, bo, out_final, nullptr, M, DMm, DMm, 0);
}
